// GLM47AttentionOptimizer_53987738911155
// MI455X (gfx1250) — matrix operand order repaired and checked by reading, not compiled, not run on hardware
//
#include <hip/hip_runtime.h>

typedef __bf16 bf16;
typedef __attribute__((ext_vector_type(16))) __bf16 v16bf;
typedef __attribute__((ext_vector_type(8)))  __bf16 v8bf;
typedef __attribute__((ext_vector_type(8)))  float   v8f;
typedef __attribute__((ext_vector_type(4)))  int     v4i;

typedef __attribute__((address_space(1))) v4i gv4i;  // global
typedef __attribute__((address_space(3))) v4i lv4i;  // LDS

#define WMMA_BF16(a, b, c) \
  __builtin_amdgcn_wmma_f32_16x16x32_bf16(false, (a), false, (b), (short)0, (c), false, false)

#if __has_builtin(__builtin_amdgcn_global_load_async_to_lds_b128)
#define HAS_ASYNC_LDS 1
#else
#define HAS_ASYNC_LDS 0
#endif

__device__ __forceinline__ v8f zero8() {
  v8f z;
#pragma unroll
  for (int i = 0; i < 8; ++i) z[i] = 0.0f;
  return z;
}

// Copy 64 bytes (32 bf16) global -> LDS. Async path uses the CDNA5
// GLOBAL_LOAD_ASYNC_TO_LDS_B128 (ASYNCcnt) data path; fallback goes via VGPRs.
__device__ __forceinline__ void stage_copy64(const bf16* gsrc, bf16* ldst) {
#if HAS_ASYNC_LDS
#pragma unroll
  for (int i = 0; i < 4; ++i)
    __builtin_amdgcn_global_load_async_to_lds_b128(
        (gv4i*)(bf16*)(gsrc + i * 8),
        (lv4i*)(ldst + i * 8), 0, 0);
#else
#pragma unroll
  for (int i = 0; i < 4; ++i)
    *(uint4*)(ldst + i * 8) = *(const uint4*)(gsrc + i * 8);
#endif
}

__device__ __forceinline__ void async_wait_all() {
#if HAS_ASYNC_LDS
  asm volatile("s_wait_asynccnt 0" ::: "memory");
#endif
}

// A fragment (16x32 bf16, MxK): lane m = lane&15; lane<16 holds K 0-7 & 16-23,
// lane>=16 holds K 8-15 & 24-31 (ISA 7.12.2). Two 16B loads.
__device__ __forceinline__ v16bf load_a_frag(const bf16* base, int stride, int kbase, int lane) {
  const int m    = lane & 15;
  const int koff = kbase + ((lane >> 4) << 3);
  const bf16* p  = base + (size_t)m * stride + koff;
  v8bf lo = *(const v8bf*)p;
  v8bf hi = *(const v8bf*)(p + 16);
  v16bf r;
#pragma unroll
  for (int i = 0; i < 8; ++i) { r[i] = lo[i]; r[i + 8] = hi[i]; }
  return r;
}

// B fragment (32x16 bf16, KxN) from source stored [N][K] row-major:
// lane n = lane&15; lane<16 reads K 0-7 & 16-23, lane>=16 reads K 8-15 & 24-31.
__device__ __forceinline__ v16bf load_b_frag(const bf16* base, int stride, int kbase, int lane) {
  const int n    = lane & 15;
  const int koff = kbase + ((lane >> 4) << 3);
  const bf16* p  = base + (size_t)n * stride + koff;
  v8bf lo = *(const v8bf*)p;
  v8bf hi = *(const v8bf*)(p + 16);
  v16bf r;
#pragma unroll
  for (int i = 0; i < 8; ++i) { r[i] = lo[i]; r[i + 8] = hi[i]; }
  return r;
}

__global__ __launch_bounds__(256) void cvt_f32_bf16(const float* __restrict__ in,
                                                    bf16* __restrict__ out, int n) {
  const int i = (blockIdx.x * 256 + threadIdx.x) * 8;
  if (i >= n) return;
  float4 a = *(const float4*)(in + i);
  float4 c = *(const float4*)(in + i + 4);
  v8bf o;
  o[0] = (bf16)a.x; o[1] = (bf16)a.y; o[2] = (bf16)a.z; o[3] = (bf16)a.w;
  o[4] = (bf16)c.x; o[5] = (bf16)c.y; o[6] = (bf16)c.z; o[7] = (bf16)c.w;
  *(v8bf*)(out + i) = o;
}

// Y = X @ W^T, X:[M,K] bf16 row-major, W:[N,K] bf16 row-major.
// Double-buffered LDS (K panel = 64), async global->LDS staging.
// EPI 0: fp32 out [M,N]   (final projection -> d_out)
// EPI 1: bf16 out head-major [B,NH,S,HD]          (Q, K)
// EPI 2: bf16 out transposed-per-head [B,NH,HD,S] (V)
template <int EPI>
__global__ __launch_bounds__(256) void gemm_bf16_wmma(const bf16* __restrict__ X,
                                                      const bf16* __restrict__ W,
                                                      void* __restrict__ out,
                                                      int M, int N, int K) {
  __shared__ bf16 As[2][128][72];  // 128 rows x 64 K (+8 pad), ping-pong
  __shared__ bf16 Bs[2][128][72];
  const int tid  = threadIdx.x;
  const int wave = tid >> 5;
  const int lane = tid & 31;
  const int ln   = lane & 15;
  const int half = lane >> 4;
  const int m0   = blockIdx.y * 128;
  const int n0   = blockIdx.x * 128;
  const int wm   = (wave & 1) * 64;   // 2 waves along M (64 rows each)
  const int wn   = (wave >> 1) * 32;  // 4 waves along N (32 cols each)

  v8f acc[4][2];
#pragma unroll
  for (int i = 0; i < 4; ++i)
#pragma unroll
    for (int j = 0; j < 2; ++j) acc[i][j] = zero8();

  const int lrow = tid >> 1;         // 0..127
  const int lseg = (tid & 1) << 5;   // 0 or 32 (each thread stages 64B per matrix)
  const bf16* xsrc = X + (size_t)(m0 + lrow) * K + lseg;
  const bf16* wsrc = W + (size_t)(n0 + lrow) * K + lseg;

  // prologue: fill stage 0
  stage_copy64(xsrc, &As[0][lrow][lseg]);
  stage_copy64(wsrc, &Bs[0][lrow][lseg]);
  async_wait_all();
  __syncthreads();

  int buf = 0;
  for (int k0 = 0; k0 < K; k0 += 64) {
    if (k0 + 64 < K) {  // issue next panel while computing this one
      stage_copy64(xsrc + k0 + 64, &As[buf ^ 1][lrow][lseg]);
      stage_copy64(wsrc + k0 + 64, &Bs[buf ^ 1][lrow][lseg]);
    }
#pragma unroll
    for (int kk = 0; kk < 64; kk += 32) {
      v16bf bfrag[2];
#pragma unroll
      for (int tn = 0; tn < 2; ++tn)
        bfrag[tn] = load_b_frag(&Bs[buf][wn + tn * 16][0], 72, kk, lane);
#pragma unroll
      for (int tm = 0; tm < 4; ++tm) {
        v16bf afrag = load_a_frag(&As[buf][wm + tm * 16][0], 72, kk, lane);
#pragma unroll
        for (int tn = 0; tn < 2; ++tn) acc[tm][tn] = WMMA_BF16(afrag, bfrag[tn], acc[tm][tn]);
      }
    }
    async_wait_all();
    __syncthreads();
    buf ^= 1;
  }

  // Epilogue. C layout: VGPR r -> row m = r + 8*half, col n = ln (ISA 7.12.2).
#pragma unroll
  for (int tm = 0; tm < 4; ++tm) {
#pragma unroll
    for (int tn = 0; tn < 2; ++tn) {
      const int n  = n0 + wn + tn * 16 + ln;
      const int mb = m0 + wm + tm * 16 + half * 8;
      if (EPI == 0) {
        float* o = (float*)out;
#pragma unroll
        for (int r = 0; r < 8; ++r) o[(size_t)(mb + r) * N + n] = acc[tm][tn][r];
      } else if (EPI == 1) {
        bf16* o = (bf16*)out;
        const int hh = n >> 7, d = n & 127;
#pragma unroll
        for (int r = 0; r < 8; ++r) {
          const int m = mb + r, b = m >> 11, s = m & 2047;
          o[(((size_t)(b * 16 + hh)) * 2048 + s) * 128 + d] = (bf16)acc[tm][tn][r];
        }
      } else {
        bf16* o = (bf16*)out;
        const int hh = n >> 7, d = n & 127;
        const int b = mb >> 11, s = mb & 2047;
        v8bf pk;
#pragma unroll
        for (int r = 0; r < 8; ++r) pk[r] = (bf16)acc[tm][tn][r];
        // 8 consecutive s values -> one 16B store
        *(v8bf*)&o[(((size_t)(b * 16 + hh)) * 128 + d) * 2048 + s] = pk;
      }
    }
  }
}

// Flash attention. Q,K: [B*NH, S, HD] bf16; Vt: [B*NH, HD, S] bf16.
// Ob: [B*S, H] bf16. One wave per 16 query rows, key blocks of 64.
__global__ __launch_bounds__(256) void flash_attn_wmma(const bf16* __restrict__ Qh,
                                                       const bf16* __restrict__ Kh,
                                                       const bf16* __restrict__ Vt,
                                                       bf16* __restrict__ Ob) {
  const int bh   = blockIdx.y;        // 0..31
  const int b    = bh >> 4, h = bh & 15;
  const int tid  = threadIdx.x;
  const int wave = tid >> 5, lane = tid & 31;
  const int ln   = lane & 15, half = lane >> 4;
  const int q0   = blockIdx.x * 128 + wave * 16;

  const bf16* Qbase = Qh + (size_t)bh * 2048 * 128 + (size_t)q0 * 128;
  const bf16* Kbase = Kh + (size_t)bh * 2048 * 128;
  const bf16* Vbase = Vt + (size_t)bh * 128 * 2048;

  __shared__ bf16 Pbuf[8][16][72];   // per-wave probs staging (C-layout -> A-layout)
  bf16(*P)[72] = Pbuf[wave];

  v16bf qf[4];
#pragma unroll
  for (int c = 0; c < 4; ++c) qf[c] = load_a_frag(Qbase, 128, 32 * c, lane);

  v8f Oacc[8];
#pragma unroll
  for (int t = 0; t < 8; ++t) Oacc[t] = zero8();
  float rmax[8], rsum[8];
#pragma unroll
  for (int r = 0; r < 8; ++r) { rmax[r] = -3.0e38f; rsum[r] = 0.0f; }

  const float scale   = 0.08838834764831845f;  // 1/sqrt(128)
  const int   nblocks = (q0 + 16 + 63) >> 6;   // causal extent for this wave

  for (int jb = 0; jb < nblocks; ++jb) {
    const int j = jb << 6;
    // ---- scores: S = Q @ K^T, 4 tiles of 16 keys ----
    v8f sacc[4];
#pragma unroll
    for (int kt = 0; kt < 4; ++kt) sacc[kt] = zero8();
#pragma unroll
    for (int kt = 0; kt < 4; ++kt) {
      const bf16* krow = Kbase + (size_t)(j + kt * 16) * 128;
#pragma unroll
      for (int c = 0; c < 4; ++c) {
        v16bf kf = load_b_frag(krow, 128, 32 * c, lane);
        sacc[kt] = WMMA_BF16(qf[c], kf, sacc[kt]);
      }
    }
    // ---- scale + causal mask + block max ----
    float bmax[8];
#pragma unroll
    for (int r = 0; r < 8; ++r) bmax[r] = -3.0e38f;
    const bool needmask = (j + 63) > q0;
#pragma unroll
    for (int kt = 0; kt < 4; ++kt) {
      const int key = j + kt * 16 + ln;
#pragma unroll
      for (int r = 0; r < 8; ++r) {
        float s = sacc[kt][r] * scale;
        if (needmask) {
          const int qrow = q0 + r + half * 8;
          if (key > qrow) s = -1.0e9f;
        }
        sacc[kt][r] = s;
        bmax[r] = fmaxf(bmax[r], s);
      }
    }
#pragma unroll
    for (int msk = 1; msk < 16; msk <<= 1)
#pragma unroll
      for (int r = 0; r < 8; ++r) bmax[r] = fmaxf(bmax[r], __shfl_xor(bmax[r], msk, 32));
    // ---- online softmax update ----
    float corr[8], bsum[8];
#pragma unroll
    for (int r = 0; r < 8; ++r) {
      const float nm = fmaxf(rmax[r], bmax[r]);
      corr[r] = __expf(rmax[r] - nm);
      rmax[r] = nm;
      bsum[r] = 0.0f;
    }
#pragma unroll
    for (int kt = 0; kt < 4; ++kt)
#pragma unroll
      for (int r = 0; r < 8; ++r) {
        const float p = __expf(sacc[kt][r] - rmax[r]);
        sacc[kt][r] = p;
        bsum[r] += p;
      }
#pragma unroll
    for (int msk = 1; msk < 16; msk <<= 1)
#pragma unroll
      for (int r = 0; r < 8; ++r) bsum[r] += __shfl_xor(bsum[r], msk, 32);
#pragma unroll
    for (int r = 0; r < 8; ++r) rsum[r] = rsum[r] * corr[r] + bsum[r];
#pragma unroll
    for (int t = 0; t < 8; ++t)
#pragma unroll
      for (int r = 0; r < 8; ++r) Oacc[t][r] *= corr[r];
    // ---- stage P (C layout) to LDS, reload as A fragments ----
#pragma unroll
    for (int kt = 0; kt < 4; ++kt)
#pragma unroll
      for (int r = 0; r < 8; ++r) P[r + half * 8][kt * 16 + ln] = (bf16)sacc[kt][r];
    asm volatile("s_wait_dscnt 0" ::: "memory");
    v16bf pf[2];
#pragma unroll
    for (int c = 0; c < 2; ++c) pf[c] = load_a_frag(&P[0][0], 72, 32 * c, lane);
    // ---- O += P @ V (V pre-transposed: rows are head-dims) ----
#pragma unroll
    for (int dt = 0; dt < 8; ++dt) {
      const bf16* vrow = Vbase + (size_t)(dt * 16) * 2048 + j;
#pragma unroll
      for (int c = 0; c < 2; ++c) {
        v16bf vf = load_b_frag(vrow, 2048, 32 * c, lane);
        Oacc[dt] = WMMA_BF16(pf[c], vf, Oacc[dt]);
      }
    }
  }
  // ---- normalize + store [B*S, H] bf16 ----
#pragma unroll
  for (int r = 0; r < 8; ++r) rsum[r] = 1.0f / rsum[r];
#pragma unroll
  for (int dt = 0; dt < 8; ++dt) {
    const int d = h * 128 + dt * 16 + ln;
#pragma unroll
    for (int r = 0; r < 8; ++r) {
      const int q = q0 + r + half * 8;
      Ob[((size_t)(b * 2048 + q)) * 2048 + d] = (bf16)(Oacc[dt][r] * rsum[r]);
    }
  }
}

extern "C" void kernel_launch(void* const* d_in, const int* in_sizes, int n_in,
                              void* d_out, int out_size, void* d_ws, size_t ws_size,
                              hipStream_t stream) {
  (void)in_sizes; (void)n_in; (void)out_size; (void)ws_size;
  const float* hidden = (const float*)d_in[0];
  // d_in[1] = attention_mask: causal, recomputed analytically in-kernel
  const float* Wq = (const float*)d_in[2];
  const float* Wk = (const float*)d_in[3];
  const float* Wv = (const float*)d_in[4];
  const float* Wo = (const float*)d_in[5];

  const int BS = 4096;  // B * S
  const int H  = 2048;

  char* ws = (char*)d_ws;
  bf16* Xb  = (bf16*)ws; ws += (size_t)BS * H * 2;
  bf16* Wqb = (bf16*)ws; ws += (size_t)H * H * 2;
  bf16* Wkb = (bf16*)ws; ws += (size_t)H * H * 2;
  bf16* Wvb = (bf16*)ws; ws += (size_t)H * H * 2;
  bf16* Wob = (bf16*)ws; ws += (size_t)H * H * 2;
  bf16* Qh  = (bf16*)ws; ws += (size_t)BS * H * 2;
  bf16* Kh  = (bf16*)ws; ws += (size_t)BS * H * 2;
  bf16* Vt  = (bf16*)ws; ws += (size_t)BS * H * 2;
  bf16* Ob  = (bf16*)ws; ws += (size_t)BS * H * 2;

  cvt_f32_bf16<<<BS * H / 8 / 256, 256, 0, stream>>>(hidden, Xb, BS * H);
  cvt_f32_bf16<<<H * H / 8 / 256, 256, 0, stream>>>(Wq, Wqb, H * H);
  cvt_f32_bf16<<<H * H / 8 / 256, 256, 0, stream>>>(Wk, Wkb, H * H);
  cvt_f32_bf16<<<H * H / 8 / 256, 256, 0, stream>>>(Wv, Wvb, H * H);
  cvt_f32_bf16<<<H * H / 8 / 256, 256, 0, stream>>>(Wo, Wob, H * H);

  dim3 g(H / 128, BS / 128);  // (16, 32)
  gemm_bf16_wmma<1><<<g, 256, 0, stream>>>(Xb, Wqb, Qh, BS, H, H);
  gemm_bf16_wmma<1><<<g, 256, 0, stream>>>(Xb, Wkb, Kh, BS, H, H);
  gemm_bf16_wmma<2><<<g, 256, 0, stream>>>(Xb, Wvb, Vt, BS, H, H);

  flash_attn_wmma<<<dim3(16, 32), 256, 0, stream>>>(Qh, Kh, Vt, Ob);

  gemm_bf16_wmma<0><<<g, 256, 0, stream>>>(Ob, Wob, d_out, BS, H, H);
}
